// SlidingWindowAttention_66984309948552
// MI455X (gfx1250) — compile-verified, matrix operand order checked by reading, not run on hardware
//
#include <hip/hip_runtime.h>
#include <hip/hip_bf16.h>

typedef __attribute__((ext_vector_type(16))) _Float16     v16h;
typedef __attribute__((ext_vector_type(8)))  _Float16     v8h;
typedef __attribute__((ext_vector_type(8)))  float        v8f;
typedef __attribute__((ext_vector_type(4)))  unsigned int u32x4;
typedef __attribute__((ext_vector_type(8)))  int          i32x8;
typedef __attribute__((ext_vector_type(4)))  int          i32x4;

#define BATCH 2
#define SEQ   4096
#define HID   2048
#define HEADS 16
#define HD    128
#define WIN   512
#define NCHK  8
#define MROWS (BATCH * SEQ)   // 8192

// ---------------------------------------------------------------------------
// Tensor Data Mover: async 2D tile load (global -> LDS) with HW padding.
// ROCm 7.2 clang-22: 5-arg builtin; amdgpu-toolchain clang-23: 6-arg.
// ---------------------------------------------------------------------------
#if __has_include(<hip/amd_detail/amd_gfx1250_TDM.h>)
#define TDM_CALL(g0, g1, g2, g3) \
    __builtin_amdgcn_tensor_load_to_lds((g0), (g1), (g2), (g3), (i32x8){}, 0)
#else
#define TDM_CALL(g0, g1, g2, g3) \
    __builtin_amdgcn_tensor_load_to_lds((g0), (g1), (g2), (g3), 0)
#endif

// Load a [rows x 32] f16 tile (row stride = rowstride elems) into LDS at
// lds_byte_off, padding +4 DWORDs after every 16 DWORDs stored so each
// 32-half row lands on a 40-half pitch (bank-conflict-free b128 frag loads).
static __device__ __forceinline__ void tdm_load_tile32(unsigned lds_byte_off,
                                                       const _Float16* gsrc,
                                                       unsigned rows,
                                                       unsigned rowstride) {
    unsigned long long ga = (unsigned long long)gsrc;
    u32x4 g0;
    g0[0] = 1u;                                   // count=1, user descriptor
    g0[1] = lds_byte_off;                         // lds_addr (bytes)
    g0[2] = (unsigned)ga;                         // global_addr[31:0]
    g0[3] = (unsigned)(ga >> 32) | (2u << 30);    // global_addr[56:32] | type=2
    i32x8 g1;
    g1[0] = (1 << 16)          // data_size = 1 (2 bytes)
          | (1 << 20)          // pad_enable
          | (3 << 22)          // pad_interval: 16 DWORDs
          | (3 << 25);         // pad_amount:   4 DWORDs
    g1[1] = (int)((rowstride & 0xFFFFu) << 16);                 // tensor_dim0 lo16
    g1[2] = (int)((rowstride >> 16) & 0xFFFFu);                 // dim0 hi | dim1 lo16(=0)
    g1[3] = (int)(((1u << 20) >> 16) | (32u << 16));            // dim1 hi16 | tile_dim0=32
    g1[4] = (int)rows;                                          // tile_dim1 | tile_dim2=0
    g1[5] = (int)rowstride;                                     // tensor_dim0_stride lo32
    g1[6] = 0;
    g1[7] = 0;
    i32x4 z = (i32x4){};                                        // groups 2/3 unused (2D)
    TDM_CALL(g0, g1, z, z);
}

// ---------------------------------------------------------------------------
// Fragment load: 16 halves for one WMMA operand.
// Lane pattern (16-bit A 16x32 / B 32x16): p = &row[hh*8]; lo=p[0..7], hi=p[16..23]
// ---------------------------------------------------------------------------
static __device__ __forceinline__ v16h load_frag(const _Float16* p) {
    v8h lo = *(const v8h*)p;
    v8h hi = *(const v8h*)(p + 16);
    v16h r;
#pragma unroll
    for (int i = 0; i < 8; ++i) { r[i] = lo[i]; r[i + 8] = hi[i]; }
    return r;
}

static __device__ __forceinline__ v8f wmma_f16(v16h a, v16h b, v8f c) {
    return __builtin_amdgcn_wmma_f32_16x16x32_f16(false, a, false, b, (short)0, c,
                                                  false, false);
}

// ---------------------------------------------------------------------------
// f32 -> f16 elementwise
// ---------------------------------------------------------------------------
__global__ __launch_bounds__(256) void cvt_f16(const float* __restrict__ in,
                                               _Float16* __restrict__ out, int n) {
    int i = blockIdx.x * 256 + threadIdx.x;
    if (i < n) out[i] = (_Float16)in[i];
}

// f32 [K,N] row-major -> f16 [N,K] row-major (transpose + convert)
__global__ __launch_bounds__(256) void cvt_trans(const float* __restrict__ W,
                                                 _Float16* __restrict__ Wt,
                                                 int K, int N) {
    int i = blockIdx.x * 256 + threadIdx.x;
    if (i < K * N) {
        int n = i % N;
        int k = i / N;
        Wt[(long)n * K + k] = (_Float16)W[i];
    }
}

// ---------------------------------------------------------------------------
// GEMM: C[M,N] = A[M,K] * B[K,N], B given transposed as Bt[N,K].
// Block tile 128x128x32, 256 threads = 8 waves (2x4), wave tile 64x32.
// Tiles staged by the Tensor Data Mover, double-buffered; wave 0 issues the
// DMA and gates it with s_wait_tensorcnt; other waves touch no memory in the
// K loop except LDS fragment loads.
// ---------------------------------------------------------------------------
template <bool F32OUT>
__global__ __launch_bounds__(256) void gemm_bt(const _Float16* __restrict__ A,
                                               const _Float16* __restrict__ Bt,
                                               void* __restrict__ Cout,
                                               int M, int Ndim, int K) {
    __shared__ __align__(16) _Float16 As[2][128][40];
    __shared__ __align__(16) _Float16 Bs[2][128][40];

    const int tid  = threadIdx.x;
    const int lane = tid & 31;
    const int wave = tid >> 5;
    const int wm   = wave >> 2;        // 0..1
    const int wn   = wave & 3;         // 0..3
    const int hh   = lane >> 4;
    const int r    = lane & 15;
    const int bm   = blockIdx.y * 128;
    const int bn   = blockIdx.x * 128;

    v8f acc[4][2];
#pragma unroll
    for (int mi = 0; mi < 4; ++mi)
#pragma unroll
        for (int ni = 0; ni < 2; ++ni) acc[mi][ni] = (v8f){};

    const int nk = K / 32;

    if (wave == 0) {
        tdm_load_tile32((unsigned)(unsigned long long)&As[0][0][0],
                        A + (long)bm * K, 128, (unsigned)K);
        tdm_load_tile32((unsigned)(unsigned long long)&Bs[0][0][0],
                        Bt + (long)bn * K, 128, (unsigned)K);
    }

    for (int it = 0; it < nk; ++it) {
        const int cur = it & 1;
        if (wave == 0) {
            if (it + 1 < nk) {
                const int nxt = (it + 1) & 1;
                tdm_load_tile32((unsigned)(unsigned long long)&As[nxt][0][0],
                                A + (long)bm * K + (it + 1) * 32, 128, (unsigned)K);
                tdm_load_tile32((unsigned)(unsigned long long)&Bs[nxt][0][0],
                                Bt + (long)bn * K + (it + 1) * 32, 128, (unsigned)K);
                __builtin_amdgcn_s_wait_tensorcnt((short)2);  // current pair done
            } else {
                __builtin_amdgcn_s_wait_tensorcnt((short)0);  // last pair done
            }
        }
        __syncthreads();   // tile[cur] visible to all waves

        v16h af[4], bf[2];
#pragma unroll
        for (int mi = 0; mi < 4; ++mi)
            af[mi] = load_frag(&As[cur][wm * 64 + mi * 16 + r][hh * 8]);
#pragma unroll
        for (int ni = 0; ni < 2; ++ni)
            bf[ni] = load_frag(&Bs[cur][wn * 32 + ni * 16 + r][hh * 8]);
#pragma unroll
        for (int mi = 0; mi < 4; ++mi)
#pragma unroll
            for (int ni = 0; ni < 2; ++ni)
                acc[mi][ni] = wmma_f16(af[mi], bf[ni], acc[mi][ni]);

        __syncthreads();   // everyone done with tile[cur] before it is re-DMA'd
    }

#pragma unroll
    for (int mi = 0; mi < 4; ++mi)
#pragma unroll
        for (int ni = 0; ni < 2; ++ni) {
            int col = bn + wn * 32 + ni * 16 + r;
#pragma unroll
            for (int v = 0; v < 8; ++v) {
                int row = bm + wm * 64 + mi * 16 + v + 8 * hh;
                if (F32OUT)
                    ((float*)Cout)[(long)row * Ndim + col] = acc[mi][ni][v];
                else
                    ((_Float16*)Cout)[(long)row * Ndim + col] =
                        (_Float16)acc[mi][ni][v];
            }
        }
}

// ---------------------------------------------------------------------------
// RoPE (+ layout change [B,N,H*D] -> [B,H,N,D]); also moves V.
// ---------------------------------------------------------------------------
__global__ __launch_bounds__(256) void rope_reorder(const _Float16* __restrict__ Qb,
                                                    const _Float16* __restrict__ Kb,
                                                    const _Float16* __restrict__ Vb,
                                                    const int* __restrict__ pos,
                                                    _Float16* __restrict__ Qr,
                                                    _Float16* __restrict__ Kr,
                                                    _Float16* __restrict__ Vr) {
    int idx = blockIdx.x * 256 + threadIdx.x;   // 2^23 total
    int i = idx & 63;
    int h = (idx >> 6) & 15;
    int n = (idx >> 10) & 4095;
    int b = idx >> 22;

    int p = pos[b * SEQ + n];
    float inv = __powf(10000.f, -(float)i * (1.0f / 64.0f));
    float ang = (float)p * inv;
    float cs = __cosf(ang), sn = __sinf(ang);

    long src = ((long)(b * SEQ + n)) * HID + h * HD + i;
    long dst = (((long)(b * HEADS + h)) * SEQ + n) * HD + i;

    float q1 = (float)Qb[src], q2 = (float)Qb[src + 64];
    Qr[dst]      = (_Float16)(q1 * cs - q2 * sn);
    Qr[dst + 64] = (_Float16)(q2 * cs + q1 * sn);

    float k1 = (float)Kb[src], k2 = (float)Kb[src + 64];
    Kr[dst]      = (_Float16)(k1 * cs - k2 * sn);
    Kr[dst + 64] = (_Float16)(k2 * cs + k1 * sn);

    Vr[dst]      = Vb[src];
    Vr[dst + 64] = Vb[src + 64];
}

// ---------------------------------------------------------------------------
// Flash-style sliding-window attention.
// grid = (W/128, nc, B*H), block = 256 (8 waves). Wave owns 16 queries.
// kv window coordinate kj in [0, 2W): key abs row = (c-1)*W + kj.
// valid iff kj > qi && kj <= qi + W (chunk 0: kj >= W via loop start).
// ---------------------------------------------------------------------------
__global__ __launch_bounds__(256) void attn_swa(const _Float16* __restrict__ Qr,
                                                const _Float16* __restrict__ Kr,
                                                const _Float16* __restrict__ Vr,
                                                _Float16* __restrict__ Ah) {
    __shared__ __align__(16) _Float16 Vt[128][40];       // V transposed: [d][kv]
    __shared__ __align__(16) _Float16 Pl[8][16][40];     // per-wave P tile [q][kv]

    const int tid  = threadIdx.x;
    const int lane = tid & 31;
    const int wave = tid >> 5;
    const int hh   = lane >> 4;
    const int r    = lane & 15;
    const int qb   = blockIdx.x;     // 0..3  (128-query slab)
    const int c    = blockIdx.y;     // chunk 0..7
    const int bh   = blockIdx.z;     // b*16+h

    const _Float16* Qp = Qr + (long)bh * SEQ * HD;
    const _Float16* Kp = Kr + (long)bh * SEQ * HD;
    const _Float16* Vp = Vr + (long)bh * SEQ * HD;

    const int qi0 = qb * 128 + wave * 16;     // wave's first query (within chunk)
    const int nq0 = c * WIN + qi0;            // absolute

    v16h qf[4];
#pragma unroll
    for (int ks = 0; ks < 4; ++ks)
        qf[ks] = load_frag(Qp + (long)(nq0 + r) * HD + ks * 32 + hh * 8);

    v8f O[8];
#pragma unroll
    for (int dt = 0; dt < 8; ++dt) O[dt] = (v8f){};
    float mrow[8], ssum[8];
#pragma unroll
    for (int v = 0; v < 8; ++v) { mrow[v] = -1e30f; ssum[v] = 0.f; }

    const int kvs  = (c == 0) ? WIN : 0;
    int kjlo = qb * 128; if (kjlo < kvs) kjlo = kvs;
    const int kjhi = qb * 128 + 608;          // last tile with any valid key
    const float scale = 0.08838834764831845f; // 1/sqrt(128)

    for (int kj0 = kjlo; kj0 <= kjhi; kj0 += 32) {
        const int nk0 = (c - 1) * WIN + kj0;  // absolute key base (>=0 by kvs)

        // prefetch next tile's K/V rows toward the WGP while we compute
        if (kj0 + 32 <= kjhi) {
            __builtin_prefetch(Kp + (long)(nk0 + 32 + r + 16 * hh) * HD, 0, 0);
            __builtin_prefetch(Vp + (long)(nk0 + 32 + (tid >> 3)) * HD, 0, 0);
        }

        // cooperative stage of V^T tile: Vt[d][kv]
        {
            int kv = tid >> 3;
            int dg = (tid & 7) * 16;
            const _Float16* gv = Vp + (long)(nk0 + kv) * HD + dg;
            v8h a = *(const v8h*)gv;
            v8h b = *(const v8h*)(gv + 8);
#pragma unroll
            for (int j = 0; j < 8; ++j) {
                Vt[dg + j][kv]     = a[j];
                Vt[dg + 8 + j][kv] = b[j];
            }
        }
        __syncthreads();

        // S = Q * K^T for 32 keys (two 16-key sub-tiles)
        v8f s0 = (v8f){}, s1 = (v8f){};
#pragma unroll
        for (int ks = 0; ks < 4; ++ks) {
            v16h kf0 = load_frag(Kp + (long)(nk0 + r) * HD + ks * 32 + hh * 8);
            s0 = wmma_f16(qf[ks], kf0, s0);
            v16h kf1 = load_frag(Kp + (long)(nk0 + 16 + r) * HD + ks * 32 + hh * 8);
            s1 = wmma_f16(qf[ks], kf1, s1);
        }

        // online softmax: rows live across the 16-lane groups of C layout
#pragma unroll
        for (int v = 0; v < 8; ++v) {
            int qi  = qi0 + v + 8 * hh;
            int kja = kj0 + r;
            int kjb = kj0 + 16 + r;
            float xa = (kja > qi && kja <= qi + WIN) ? s0[v] * scale : -3e38f;
            float xb = (kjb > qi && kjb <= qi + WIN) ? s1[v] * scale : -3e38f;
            float rm = fmaxf(xa, xb);
            rm = fmaxf(rm, __shfl_xor(rm, 1, 32));
            rm = fmaxf(rm, __shfl_xor(rm, 2, 32));
            rm = fmaxf(rm, __shfl_xor(rm, 4, 32));
            rm = fmaxf(rm, __shfl_xor(rm, 8, 32));
            float mc   = fmaxf(mrow[v], rm);
            float corr = __expf(mrow[v] - mc);
            float pa   = __expf(xa - mc);
            float pb   = __expf(xb - mc);
            float ps   = pa + pb;
            ps += __shfl_xor(ps, 1, 32);
            ps += __shfl_xor(ps, 2, 32);
            ps += __shfl_xor(ps, 4, 32);
            ps += __shfl_xor(ps, 8, 32);
            ssum[v] = ssum[v] * corr + ps;
            mrow[v] = mc;
#pragma unroll
            for (int dt = 0; dt < 8; ++dt) O[dt][v] = O[dt][v] * corr;
            Pl[wave][v + 8 * hh][r]      = (_Float16)pa;
            Pl[wave][v + 8 * hh][16 + r] = (_Float16)pb;
        }

        // O += P * V   (P: A-matrix 16x32 from per-wave LDS; V^T: B-matrix)
        v16h pf = load_frag(&Pl[wave][r][hh * 8]);
#pragma unroll
        for (int dt = 0; dt < 8; ++dt) {
            v16h vf = load_frag(&Vt[dt * 16 + r][hh * 8]);
            O[dt] = wmma_f16(pf, vf, O[dt]);
        }
        __syncthreads();
    }

    // epilogue: normalize and write [B,N,H*D] f16 for the output projection
    const int b = bh >> 4, h = bh & 15;
#pragma unroll
    for (int v = 0; v < 8; ++v) {
        float rcp = 1.0f / ssum[v];
        int nrow = c * WIN + qi0 + v + 8 * hh;
        long base = ((long)(b * SEQ + nrow)) * HID + h * HD;
#pragma unroll
        for (int dt = 0; dt < 8; ++dt)
            Ah[base + dt * 16 + r] = (_Float16)(O[dt][v] * rcp);
    }
}

// ---------------------------------------------------------------------------
// launch
// ---------------------------------------------------------------------------
extern "C" void kernel_launch(void* const* d_in, const int* in_sizes, int n_in,
                              void* d_out, int out_size, void* d_ws, size_t ws_size,
                              hipStream_t stream) {
    const float* x   = (const float*)d_in[0];
    const int*   pos = (const int*)d_in[1];
    const float* Wq  = (const float*)d_in[2];
    const float* Wk  = (const float*)d_in[3];
    const float* Wv  = (const float*)d_in[4];
    const float* Wo  = (const float*)d_in[5];
    float* out = (float*)d_out;

    const long XE = (long)MROWS * HID;       // 16,777,216
    const long WE = (long)HID * HID;         // 4,194,304

    _Float16* p   = (_Float16*)d_ws;
    _Float16* xh  = p; p += XE;
    _Float16* Wqh = p; p += WE;
    _Float16* Wkh = p; p += WE;
    _Float16* Wvh = p; p += WE;
    _Float16* Woh = p; p += WE;
    _Float16* Qb  = p; p += XE;
    _Float16* Kb  = p; p += XE;
    _Float16* Vb  = p; p += XE;
    _Float16* Qr  = p; p += XE;
    _Float16* Kr  = p; p += XE;
    _Float16* Vr  = p; p += XE;
    _Float16* Ah  = Qb;                      // reuse Q projection buffer

    cvt_f16<<<(int)(XE / 256), 256, 0, stream>>>(x, xh, (int)XE);
    cvt_trans<<<(int)(WE / 256), 256, 0, stream>>>(Wq, Wqh, HID, HID);
    cvt_trans<<<(int)(WE / 256), 256, 0, stream>>>(Wk, Wkh, HID, HID);
    cvt_trans<<<(int)(WE / 256), 256, 0, stream>>>(Wv, Wvh, HID, HID);
    cvt_trans<<<(int)(WE / 256), 256, 0, stream>>>(Wo, Woh, HID, HID);

    dim3 gg(HID / 128, MROWS / 128);         // (16, 64)
    gemm_bt<false><<<gg, 256, 0, stream>>>(xh, Wqh, Qb, MROWS, HID, HID);
    gemm_bt<false><<<gg, 256, 0, stream>>>(xh, Wkh, Kb, MROWS, HID, HID);
    gemm_bt<false><<<gg, 256, 0, stream>>>(xh, Wvh, Vb, MROWS, HID, HID);

    rope_reorder<<<(BATCH * SEQ * HEADS * 64) / 256, 256, 0, stream>>>(
        Qb, Kb, Vb, pos, Qr, Kr, Vr);

    attn_swa<<<dim3(WIN / 128, NCHK, BATCH * HEADS), 256, 0, stream>>>(Qr, Kr, Vr, Ah);

    gemm_bt<true><<<gg, 256, 0, stream>>>(Ah, Woh, out, MROWS, HID, HID);
}
